// LSTM_50964081935314
// MI455X (gfx1250) — matrix operand order repaired and checked by reading, not compiled, not run on hardware
//
#include <hip/hip_runtime.h>

// ---------------------------------------------------------------------------
// LSTM on MI455X (gfx1250, wave32, WMMA).
// Fused persistent kernel: z_t = [x_t, h_t] @ [Wx; Wh] + b via
// v_wmma_f32_16x16x32_bf16. Cell state + W_h fragments live in VGPRs for all
// 1024 steps, W_x fragments live in LDS. Software-pipelined: the x-part GEMM
// for step t+1 executes inside the split device barrier (arrive / wait), so
// the serial critical path is only  barrier -> h-GEMM -> gates -> h store.
// ---------------------------------------------------------------------------

typedef __bf16  v16bf __attribute__((ext_vector_type(16)));
typedef float   v8f   __attribute__((ext_vector_type(8)));
typedef float   fv4   __attribute__((ext_vector_type(4)));
typedef unsigned int uv4 __attribute__((ext_vector_type(4)));

#define B_   128
#define T_   1024
#define D_   256
#define H_   256
#define C_   128
#define NWG  16      // one block per n-tile of H
#define NWAVES 8     // one wave per m-tile of B

// ws layout (bytes)
#define OFF_CNT   0
#define OFF_HPACK 256                          // bf16 [2][8 mt][8 kt][32 lane][16] = 131072 B (262144 B reserved)
#define OFF_WXP   (OFF_HPACK + 262144)         // bf16 [4][8 kt][16 nt][32][16]    = 524288 B
#define OFF_WHP   (OFF_WXP + 524288)           // bf16 [4][8 kt][16 nt][32][16]    = 524288 B
#define OFF_WHYP  (OFF_WHP + 524288)           // bf16 [8 kt][8 nt][32][16]        = 65536 B

static __device__ __forceinline__ v16bf ldfrag(const unsigned short* __restrict__ p) {
  union { v16bf v; uv4 q[2]; } u;
  const uv4* q = (const uv4*)p;   // 32B-aligned by construction
  u.q[0] = q[0];
  u.q[1] = q[1];
  return u.v;
}

static __device__ __forceinline__ v16bf ldfrag_lds(const unsigned short* p) {
  union { v16bf v; uv4 q[2]; } u;
  const uv4* q = (const uv4*)p;   // LDS, 16B-aligned -> ds_load_b128 x2
  u.q[0] = q[0];
  u.q[1] = q[1];
  return u.v;
}

static __device__ __forceinline__ v8f wmma_bf16(v16bf a, v16bf b, v8f c) {
  return __builtin_amdgcn_wmma_f32_16x16x32_bf16(false, a, false, b, (short)0, c,
                                                 false, false);
}

// fast sigmoid / tanh: v_exp_f32 + v_rcp_f32 (precision >> bf16 matmul noise)
static __device__ __forceinline__ float sigm(float z) {
  return __builtin_amdgcn_rcpf(1.0f + __expf(-z));
}
static __device__ __forceinline__ float tanh_fast(float z) {
  float e = __expf(2.0f * z);
  return 1.0f - 2.0f * __builtin_amdgcn_rcpf(e + 1.0f);
}

static __device__ __forceinline__ unsigned short f2bf(float f) {
  return __builtin_bit_cast(unsigned short, (__bf16)f);
}

// x-part GEMM for one timestep: zx[g] = x_row_tile @ Wx_g (K = D = 256).
// A fragments built from f32 x with packed bf16 converts; B fragments from LDS.
static __device__ __forceinline__ void compute_zx(v8f zx[4],
                                                  const float* __restrict__ xrow,
                                                  const unsigned short* lds_wx,
                                                  int lane, int hi) {
#pragma unroll
  for (int g = 0; g < 4; ++g) zx[g] = (v8f){0.f, 0.f, 0.f, 0.f, 0.f, 0.f, 0.f, 0.f};
#pragma unroll
  for (int kt = 0; kt < 8; ++kt) {
    const int kb = kt * 32 + hi * 8;
    float tmp[16];
    *(fv4*)(tmp + 0)  = *(const fv4*)(xrow + kb);
    *(fv4*)(tmp + 4)  = *(const fv4*)(xrow + kb + 4);
    *(fv4*)(tmp + 8)  = *(const fv4*)(xrow + kb + 16);
    *(fv4*)(tmp + 12) = *(const fv4*)(xrow + kb + 20);
    v16bf af;
#pragma unroll
    for (int e = 0; e < 16; ++e) af[e] = (__bf16)tmp[e];
    // batch the 4 gate B-fragment LDS loads ahead of the 4 WMMAs
    v16bf b0 = ldfrag_lds(lds_wx + ((0 * 8 + kt) * 32 + lane) * 16);
    v16bf b1 = ldfrag_lds(lds_wx + ((1 * 8 + kt) * 32 + lane) * 16);
    v16bf b2 = ldfrag_lds(lds_wx + ((2 * 8 + kt) * 32 + lane) * 16);
    v16bf b3 = ldfrag_lds(lds_wx + ((3 * 8 + kt) * 32 + lane) * 16);
    zx[0] = wmma_bf16(af, b0, zx[0]);
    zx[1] = wmma_bf16(af, b1, zx[1]);
    zx[2] = wmma_bf16(af, b2, zx[2]);
    zx[3] = wmma_bf16(af, b3, zx[3]);
  }
}

// Pack a row-major [K x N] f32 weight matrix into CDNA5 WMMA B-fragment
// layout as bf16: dst[((kt*(N/16)+nt)*32+lane)*16+j], where the element at
// (lane, j) is W[(kt*32 + (j>=8?16:0) + (lane>=16?8:0) + (j&7)) * N + nt*16 + (lane&15)].
__global__ void lstm_pack_b(const float* __restrict__ W,
                            unsigned short* __restrict__ dst, int K, int N) {
  int idx = blockIdx.x * blockDim.x + threadIdx.x;
  int total = K * N;
  if (idx >= total) return;
  int j    = idx & 15;
  int lane = (idx >> 4) & 31;
  int rest = idx >> 9;
  int ntiles = N >> 4;
  int nt = rest % ntiles;
  int kt = rest / ntiles;
  int k = kt * 32 + ((j & 8) << 1) + ((lane >> 4) << 3) + (j & 7);
  int n = nt * 16 + (lane & 15);
  dst[idx] = f2bf(W[(size_t)k * N + n]);
}

__global__ void lstm_init(unsigned* __restrict__ hp32, unsigned* __restrict__ cnt, int n) {
  int i = blockIdx.x * blockDim.x + threadIdx.x;
  if (i == 0) *cnt = 0;
  if (i < n) hp32[i] = 0u;
}

__global__ __launch_bounds__(256, 1)
void lstm_persistent(const float* __restrict__ x,        // [B,T,D]
                     const float* __restrict__ b_i, const float* __restrict__ b_f,
                     const float* __restrict__ b_g, const float* __restrict__ b_o,
                     const float* __restrict__ b_y,
                     const unsigned short* __restrict__ wxp,   // [4][8][16][32][16]
                     const unsigned short* __restrict__ whp,   // [4][8][16][32][16]
                     const unsigned short* __restrict__ whyp,  // [8][8][32][16]
                     unsigned short* __restrict__ hpack,       // [2][8][8][32][16]
                     unsigned* __restrict__ cnt,
                     float* __restrict__ out)                  // [B,C]
{
  const int nt   = blockIdx.x;            // n-tile of H: 0..15
  const int mt   = threadIdx.x >> 5;      // m-tile of B: 0..7 (wave id)
  const int lane = threadIdx.x & 31;
  const int nloc = lane & 15;
  const int hi   = lane >> 4;             // K-half selector in A layout

  // Stage this WG's W_x B-fragments in LDS (loop-invariant, 32 KB of 320 KB).
  __shared__ unsigned short lds_wx[4 * 8 * 32 * 16];   // 32 KB
  {
    const int tid = threadIdx.x;
#pragma unroll
    for (int i = 0; i < 8; ++i) {                       // 2048 uv4 / 256 thr
      int q   = tid + i * 256;                          // uv4 index
      int blk = q >> 6;                                 // (g,kt): 64 uv4/blk
      int off = q & 63;
      int g   = blk >> 3;
      int kt  = blk & 7;
      *(uv4*)(lds_wx + (size_t)blk * 512 + off * 8) =
          *(const uv4*)(wxp + (size_t)g * (D_ * H_) + (kt * 16 + nt) * 512 + off * 8);
    }
  }

  // W_h B-fragments stay in VGPRs: the recurrent matmul is the latency-
  // critical path; its B operands must never touch memory.
  v16bf whf[4][8];
#pragma unroll
  for (int g = 0; g < 4; ++g)
#pragma unroll
    for (int kt = 0; kt < 8; ++kt)
      whf[g][kt] = ldfrag(whp + (size_t)g * (D_ * H_) + ((kt * 16 + nt) * 32 + lane) * 16);

  const float biv = b_i[nt * 16 + nloc];
  const float bfv = b_f[nt * 16 + nloc];
  const float bgv = b_g[nt * 16 + nloc];
  const float bov = b_o[nt * 16 + nloc];

  v8f c = {0.f, 0.f, 0.f, 0.f, 0.f, 0.f, 0.f, 0.f};   // cell state, lives in VGPRs

  const float* xrow_base = x + (size_t)(mt * 16 + nloc) * T_ * D_;  // A row = lane%16

  __syncthreads();   // LDS staging visible to all waves

  // Pipeline prologue: x-part for t = 0.
  v8f zx[4];
  compute_zx(zx, xrow_base, lds_wx, lane, hi);

  for (int t = 0; t < T_; ++t) {
    const int bufr = t & 1;
    const int bufw = bufr ^ 1;

    v8f z[4];
#pragma unroll
    for (int g = 0; g < 4; ++g) z[g] = zx[g];

    // ---- h part: z += h_t @ Wh (K = H = 256). Batch all 8 A-fragment
    // loads into one clause so L2 latency is paid once, not 8 times. ----
    v16bf ah[8];
#pragma unroll
    for (int kt = 0; kt < 8; ++kt)
      ah[kt] = ldfrag(hpack + ((((size_t)bufr * 8 + mt) * 8 + kt) * 32 + lane) * 16);
#pragma unroll
    for (int kt = 0; kt < 8; ++kt)
#pragma unroll
      for (int g = 0; g < 4; ++g) z[g] = wmma_bf16(ah[kt], whf[g][kt], z[g]);

    // ---- gates + cell update; scatter h_new into A-fragment layout ----
#pragma unroll
    for (int v = 0; v < 8; ++v) {
      float iv = sigm(z[0][v] + biv);
      float fv = sigm(z[1][v] + bfv);
      float gv = tanh_fast(z[2][v] + bgv);
      float ov = sigm(z[3][v] + bov);
      float cv = fv * c[v] + iv * gv;
      c[v] = cv;
      float hv = ov * tanh_fast(cv);
      // D-layout position: row r = v + hi*8 (b), col nloc (h). Next-step A
      // fragment position for (row r, k = nt*16+nloc):
      int r     = v + hi * 8;
      int lanep = r + ((nloc >= 8) ? 16 : 0);
      int j     = ((nt & 1) << 3) + (nloc & 7);
      hpack[((((size_t)bufw * 8 + mt) * 8 + (nt >> 1)) * 32 + lanep) * 16 + j] = f2bf(hv);
    }

    // ---- split device barrier with overlapped x-part for t+1 ----
    __threadfence();                 // release h stores (agent scope)
    __syncthreads();                 // all waves in WG arrived
    if (threadIdx.x == 0) {
      __hip_atomic_fetch_add(cnt, 1u, __ATOMIC_RELAXED, __HIP_MEMORY_SCOPE_AGENT);
    }
    if (t + 1 < T_) {                // off the critical path: overlap with barrier
      const float* xrow = xrow_base + (size_t)(t + 1) * D_;
      if (t + 2 < T_) __builtin_prefetch(xrow + D_, 0, 3);
      compute_zx(zx, xrow, lds_wx, lane, hi);
    }
    if (threadIdx.x == 0) {
      unsigned target = (unsigned)(NWG * (t + 1));
      while (__hip_atomic_load(cnt, __ATOMIC_RELAXED, __HIP_MEMORY_SCOPE_AGENT) < target) {
        __builtin_amdgcn_s_sleep(2);
      }
    }
    __syncthreads();                 // barrier complete for whole WG
    __threadfence();                 // acquire: invalidate near caches
  }

  // ---- out = h_T @ W_hy + b_y  (h_T is in hpack buffer 0) ----
  if (nt < 8) {
    v8f acc = {0.f, 0.f, 0.f, 0.f, 0.f, 0.f, 0.f, 0.f};
#pragma unroll
    for (int kt = 0; kt < 8; ++kt) {
      v16bf af    = ldfrag(hpack + (((size_t)mt * 8 + kt) * 32 + lane) * 16);
      v16bf bfrag = ldfrag(whyp + ((kt * 8 + nt) * 32 + lane) * 16);
      acc = wmma_bf16(af, bfrag, acc);
    }
    float by = b_y[nt * 16 + nloc];
#pragma unroll
    for (int v = 0; v < 8; ++v) {
      int r = v + hi * 8;
      out[(size_t)(mt * 16 + r) * C_ + nt * 16 + nloc] = acc[v] + by;
    }
  }
}

extern "C" void kernel_launch(void* const* d_in, const int* in_sizes, int n_in,
                              void* d_out, int out_size, void* d_ws, size_t ws_size,
                              hipStream_t stream) {
  (void)in_sizes; (void)n_in; (void)out_size; (void)ws_size;

  const float* x    = (const float*)d_in[0];
  const float* W_xi = (const float*)d_in[1];
  const float* W_hi = (const float*)d_in[2];
  const float* b_i  = (const float*)d_in[3];
  const float* W_xf = (const float*)d_in[4];
  const float* W_hf = (const float*)d_in[5];
  const float* b_f  = (const float*)d_in[6];
  const float* W_xg = (const float*)d_in[7];
  const float* W_hg = (const float*)d_in[8];
  const float* b_g  = (const float*)d_in[9];
  const float* W_xo = (const float*)d_in[10];
  const float* W_ho = (const float*)d_in[11];
  const float* b_o  = (const float*)d_in[12];
  const float* W_hy = (const float*)d_in[13];
  const float* b_y  = (const float*)d_in[14];

  char* ws = (char*)d_ws;
  unsigned*       cnt   = (unsigned*)(ws + OFF_CNT);
  unsigned short* hpack = (unsigned short*)(ws + OFF_HPACK);
  unsigned short* wxp   = (unsigned short*)(ws + OFF_WXP);
  unsigned short* whp   = (unsigned short*)(ws + OFF_WHP);
  unsigned short* whyp  = (unsigned short*)(ws + OFF_WHYP);

  const int nW  = D_ * H_;           // 65536 elements per gate matrix
  const int nWy = H_ * C_;           // 32768
  dim3 pb(256);
  dim3 pgW((nW + 255) / 256), pgWy((nWy + 255) / 256);

  lstm_pack_b<<<pgW, pb, 0, stream>>>(W_xi, wxp + 0 * (size_t)nW, D_, H_);
  lstm_pack_b<<<pgW, pb, 0, stream>>>(W_xf, wxp + 1 * (size_t)nW, D_, H_);
  lstm_pack_b<<<pgW, pb, 0, stream>>>(W_xg, wxp + 2 * (size_t)nW, D_, H_);
  lstm_pack_b<<<pgW, pb, 0, stream>>>(W_xo, wxp + 3 * (size_t)nW, D_, H_);
  lstm_pack_b<<<pgW, pb, 0, stream>>>(W_hi, whp + 0 * (size_t)nW, H_, H_);
  lstm_pack_b<<<pgW, pb, 0, stream>>>(W_hf, whp + 1 * (size_t)nW, H_, H_);
  lstm_pack_b<<<pgW, pb, 0, stream>>>(W_hg, whp + 2 * (size_t)nW, H_, H_);
  lstm_pack_b<<<pgW, pb, 0, stream>>>(W_ho, whp + 3 * (size_t)nW, H_, H_);
  lstm_pack_b<<<pgWy, pb, 0, stream>>>(W_hy, whyp, H_, C_);

  // zero both h buffers (h0 = 0) + reset barrier counter every launch
  lstm_init<<<(65536 + 255) / 256, 256, 0, stream>>>((unsigned*)hpack, cnt, 65536);

  lstm_persistent<<<dim3(NWG), dim3(NWAVES * 32), 0, stream>>>(
      x, b_i, b_f, b_g, b_o, b_y, wxp, whp, whyp, hpack, cnt, (float*)d_out);
}
